// WindowAttention_82300163326493
// MI455X (gfx1250) — matrix operand order repaired and checked by reading, not compiled, not run on hardware
//
#include <hip/hip_runtime.h>

typedef __attribute__((ext_vector_type(16))) _Float16 v16h;
typedef __attribute__((ext_vector_type(8)))  _Float16 v8h;
typedef __attribute__((ext_vector_type(8)))  float    v8f;

constexpr int kNW   = 8192;
constexpr int kN    = 49;    // tokens per window
constexpr int kNP   = 64;    // padded tokens
constexpr int kC    = 128;   // channels
constexpr int kH    = 4;     // heads
constexpr int kD    = 32;    // head dim

// ---- WMMA fragment loaders (per CDNA5 ISA VGPR layouts) ----

// A: 16x32 f16, row-major with leading dim `ld` (halves).
// lane 0-15:  M=lane,    K = 0..7 (v0-3) and 16..23 (v4-7)
// lane 16-31: M=lane-16, K = 8..15       and 24..31
__device__ __forceinline__ v16h load_a_f16(const _Float16* base, int ld) {
  const int lane  = threadIdx.x & 31;
  const _Float16* row = base + (lane & 15) * ld + ((lane >> 4) << 3);
  v8h lo = *(const v8h*)(row);
  v8h hi = *(const v8h*)(row + 16);
  v16h a;
#pragma unroll
  for (int i = 0; i < 8; ++i) { a[i] = lo[i]; a[i + 8] = hi[i]; }
  return a;
}

// B: 32x16 f16 where element (k,n) lives at base[n*ld + k]  ("N-major")
// lane 0-15: N=lane, K=0..7 and 16..23 ; lane 16-31: N=lane-16, K=8..15 and 24..31
// -> two 16B runs per lane.
__device__ __forceinline__ v16h load_b_nm(const _Float16* base, int ld) {
  const int lane = threadIdx.x & 31;
  const _Float16* p = base + (lane & 15) * ld + ((lane >> 4) << 3);
  v8h lo = *(const v8h*)(p);
  v8h hi = *(const v8h*)(p + 16);
  v16h b;
#pragma unroll
  for (int i = 0; i < 8; ++i) { b[i] = lo[i]; b[i + 8] = hi[i]; }
  return b;
}

#define WMMA_F16(A, B, C) \
  __builtin_amdgcn_wmma_f32_16x16x32_f16(false, (A), false, (B), (short)0, (C), false, false)

// K=128 dot: 4 chained WMMAs, A fragments streamed, B fragments pre-held.
__device__ __forceinline__ v8f mm_k128(const _Float16* abase, int ld,
                                       v16h b0, v16h b1, v16h b2, v16h b3) {
  v8f c = {};
  c = WMMA_F16(load_a_f16(abase + 0 * 32, ld), b0, c);
  c = WMMA_F16(load_a_f16(abase + 1 * 32, ld), b1, c);
  c = WMMA_F16(load_a_f16(abase + 2 * 32, ld), b2, c);
  c = WMMA_F16(load_a_f16(abase + 3 * 32, ld), b3, c);
  return c;
}

// ---- prep: f32->f16 weights, expand relative-position bias to (H,49,49) ----
__global__ __launch_bounds__(256) void prep_kernel(
    const float* __restrict__ Wqkv, const float* __restrict__ Wproj,
    const float* __restrict__ btab, const int* __restrict__ ridx,
    _Float16* __restrict__ wq16, _Float16* __restrict__ wp16,
    float* __restrict__ biasH) {
  int i = blockIdx.x * 256 + threadIdx.x;
  if (i < 3 * kC * kC) wq16[i] = (_Float16)Wqkv[i];
  if (i < kC * kC)     wp16[i] = (_Float16)Wproj[i];
  if (i < kH * kN * kN) {
    int h = i / (kN * kN);
    int nm = i % (kN * kN);
    biasH[i] = btab[ridx[nm] * kH + h];
  }
}

// ---- fused per-window attention: 1 block = 1 window, 8 waves ----
// LDS (128 KB -> 2 blocks/WGP):
//   [0    KB) x16  [tok][c] f16 (16KB)   -- reused as o16 [tok][c] (D/E)
//   [16   KB) q16  [h][tok][d]  (16KB) \__ reused as p16 [h][n][m] (32KB)
//   [32   KB) k16  [h][tok][d]  (16KB) /
//   [48   KB) vT16 [h][d][tok]  (16KB)
//   [64   KB) s32  [h][n][m] f32 (64KB)
__global__ __launch_bounds__(256) void win_attn_kernel(
    const float* __restrict__ x, const float* __restrict__ mask,
    const float* __restrict__ bqkv, const float* __restrict__ bproj,
    const _Float16* __restrict__ wq16, const _Float16* __restrict__ wp16,
    const float* __restrict__ biasH, float* __restrict__ out) {
  extern __shared__ char smem[];
  _Float16* x16  = (_Float16*)smem;
  _Float16* q16  = x16 + kNP * kC;
  _Float16* k16  = q16 + kH * kNP * kD;
  _Float16* vT16 = k16 + kH * kNP * kD;
  float*    s32  = (float*)(vT16 + kH * kD * kNP);
  _Float16* p16  = q16;   // aliases q+k (dead after phase C)
  _Float16* o16  = x16;   // aliases x  (dead after phase B)

  const int w     = blockIdx.x;
  const int tid   = threadIdx.x;
  const int wid   = tid >> 5;
  const int lane  = tid & 31;
  const int n16   = lane & 15;
  const int mbase = (lane >> 4) << 3;
  const float scale = 0.17677669529663687f; // 1/sqrt(32)

  // Phase A: stage x as f16, zero-pad tokens 49..63
  for (int i = tid; i < kNP * kC; i += 256) {
    int tok = i >> 7, cc = i & 127;
    x16[i] = (tok < kN) ? (_Float16)x[(size_t)w * kN * kC + tok * kC + cc]
                        : (_Float16)0.0f;
  }
  __syncthreads();

  // ---- Phase B: qkv = x * Wqkv^T + bqkv (M=64,N=384,K=128) ----
  // ot = wid+8i; segment (q/k/v) is uniform per unrolled section -> no branches.
  // Segment Q: ot = wid, o in [0,128)
  {
    const _Float16* wb = wq16 + wid * 16 * kC;
    v16h b0 = load_b_nm(wb + 0 * 32, kC), b1 = load_b_nm(wb + 1 * 32, kC);
    v16h b2 = load_b_nm(wb + 2 * 32, kC), b3 = load_b_nm(wb + 3 * 32, kC);
    int o = wid * 16 + n16;
    float bq = bqkv[o];
    _Float16* dst = q16 + (o >> 5) * kNP * kD + (o & 31);
#pragma unroll
    for (int mt = 0; mt < 4; ++mt) {
      v8f c = mm_k128(x16 + mt * 16 * kC, kC, b0, b1, b2, b3);
#pragma unroll
      for (int r = 0; r < 8; ++r)
        dst[(mt * 16 + mbase + r) * kD] = (_Float16)((c[r] + bq) * scale);
    }
  }
  // Segment K: ot = wid+8, o in [128,256)
  {
    const _Float16* wb = wq16 + (wid + 8) * 16 * kC;
    v16h b0 = load_b_nm(wb + 0 * 32, kC), b1 = load_b_nm(wb + 1 * 32, kC);
    v16h b2 = load_b_nm(wb + 2 * 32, kC), b3 = load_b_nm(wb + 3 * 32, kC);
    int oo = wid * 16 + n16;
    float bq = bqkv[128 + oo];
    _Float16* dst = k16 + (oo >> 5) * kNP * kD + (oo & 31);
#pragma unroll
    for (int mt = 0; mt < 4; ++mt) {
      v8f c = mm_k128(x16 + mt * 16 * kC, kC, b0, b1, b2, b3);
#pragma unroll
      for (int r = 0; r < 8; ++r)
        dst[(mt * 16 + mbase + r) * kD] = (_Float16)(c[r] + bq);
    }
  }
  // Segment V: ot = wid+16, o in [256,384); vT layout [h][d][tok] ->
  // token (M) innermost: one contiguous v8h store per lane.
  {
    const _Float16* wb = wq16 + (wid + 16) * 16 * kC;
    v16h b0 = load_b_nm(wb + 0 * 32, kC), b1 = load_b_nm(wb + 1 * 32, kC);
    v16h b2 = load_b_nm(wb + 2 * 32, kC), b3 = load_b_nm(wb + 3 * 32, kC);
    int oo = wid * 16 + n16;
    float bq = bqkv[256 + oo];
    _Float16* dst = vT16 + (oo >> 5) * kD * kNP + (oo & 31) * kNP;
#pragma unroll
    for (int mt = 0; mt < 4; ++mt) {
      v8f c = mm_k128(x16 + mt * 16 * kC, kC, b0, b1, b2, b3);
      v8h ov;
#pragma unroll
      for (int r = 0; r < 8; ++r) ov[r] = (_Float16)(c[r] + bq);
      *(v8h*)(dst + mt * 16 + mbase) = ov;
    }
  }
  __syncthreads();

  // ---- Phase C: S^T tiles = k . q^T (A=k rows=key m, B=q^T) ----
  // Tile rows M=m (innermost of s32[h][n][m]) -> contiguous 32B stores;
  // softmax rows stay contiguous too.
  for (int i = 0; i < 8; ++i) {
    int t = wid + (i << 3);
    int h = t >> 4, rem = t & 15;
    int mt = rem >> 2, nt = rem & 3;      // mt: key tile, nt: query tile
    v16h a = load_a_f16(k16 + h * kNP * kD + mt * 16 * kD, kD);
    v16h b = load_b_nm(q16 + h * kNP * kD + nt * 16 * kD, kD); // (k=d,n=q)=q[n][d]
    v8f c = {};
    c = WMMA_F16(a, b, c);
    float* sdst = s32 + h * kNP * kNP + (nt * 16 + n16) * kNP + mt * 16 + mbase;
#pragma unroll
    for (int r = 0; r < 8; ++r) sdst[r] = c[r];
  }
  __syncthreads();

  // ---- Softmax over valid 49 keys, + rel-pos bias + window mask ----
  if (tid < kH * kN) {
    int h = tid / kN, n = tid % kN;
    float* srow = s32 + h * kNP * kNP + n * kNP;
    const float* brow = biasH + h * kN * kN + n * kN;
    const float* mrow = mask + (size_t)w * kN * kN + n * kN;
    float mx = -1e30f;
    for (int m = 0; m < kN; ++m) {
      float v = srow[m] + brow[m] + mrow[m];
      srow[m] = v;
      mx = fmaxf(mx, v);
    }
    float sum = 0.f;
    for (int m = 0; m < kN; ++m) {
      float e = __expf(srow[m] - mx);
      srow[m] = e;
      sum += e;
    }
    float inv = 1.f / sum;
    _Float16* prow = p16 + h * kNP * kNP + n * kNP;
    for (int m = 0; m < kN; ++m) prow[m] = (_Float16)(srow[m] * inv);
    for (int m = kN; m < kNP; ++m) prow[m] = (_Float16)0.0f; // K-dim padding
  }
  __syncthreads();

  // ---- Phase D: O^T tiles = vT . P^T (A=vT rows=d, B=P^T) ----
  // Stored back transposed into row-major o16[tok][c]: M=d innermost there,
  // so each lane writes one contiguous v8h.
  for (int i = 0; i < 4; ++i) {
    int t = wid + (i << 3);               // 0..31
    int h = t >> 3, rem = t & 7;
    int mt = rem >> 2, nt = rem & 3;      // mt: d tile (0..1), nt: query tile
    v8f c = {};
#pragma unroll
    for (int kk = 0; kk < 2; ++kk) {
      v16h a = load_a_f16(vT16 + h * kD * kNP + mt * 16 * kNP + kk * 32, kNP);
      v16h b = load_b_nm(p16 + h * kNP * kNP + (nt * 16) * kNP + kk * 32, kNP);
      c = WMMA_F16(a, b, c);
    }
    v8h ov;
#pragma unroll
    for (int r = 0; r < 8; ++r) ov[r] = (_Float16)c[r];
    *(v8h*)(o16 + (nt * 16 + n16) * kC + h * kD + mt * 16 + mbase) = ov;
  }
  __syncthreads();

  // ---- Phase E: out^T tiles = Wproj . O^T (A=Wproj rows=o, B=O^T) ----
  // M=o innermost in out[w][tok][o] -> contiguous 32B coalesced global stores.
  // One A-fragment set (weights) per wave, reused over all 4 token tiles.
  {
    int mt = wid;                               // output-channel tile 0..7
    const _Float16* wa = wp16 + mt * 16 * kC;
    v16h a0 = load_a_f16(wa + 0 * 32, kC);
    v16h a1 = load_a_f16(wa + 1 * 32, kC);
    v16h a2 = load_a_f16(wa + 2 * 32, kC);
    v16h a3 = load_a_f16(wa + 3 * 32, kC);
    const float* bpp = bproj + mt * 16 + mbase; // 8 contiguous channels
#pragma unroll
    for (int nt = 0; nt < 4; ++nt) {
      v8f c = {};
      c = WMMA_F16(a0, load_b_nm(o16 + (nt * 16) * kC + 0 * 32, kC), c);
      c = WMMA_F16(a1, load_b_nm(o16 + (nt * 16) * kC + 1 * 32, kC), c);
      c = WMMA_F16(a2, load_b_nm(o16 + (nt * 16) * kC + 2 * 32, kC), c);
      c = WMMA_F16(a3, load_b_nm(o16 + (nt * 16) * kC + 3 * 32, kC), c);
      int tok = nt * 16 + n16;
      if (tok < kN) {
        float* od = out + (size_t)w * kN * kC + tok * kC + mt * 16 + mbase;
#pragma unroll
        for (int r = 0; r < 8; ++r) od[r] = c[r] + bpp[r];
      }
    }
  }
}

extern "C" void kernel_launch(void* const* d_in, const int* in_sizes, int n_in,
                              void* d_out, int out_size, void* d_ws, size_t ws_size,
                              hipStream_t stream) {
  (void)in_sizes; (void)n_in; (void)out_size; (void)ws_size;
  const float* x     = (const float*)d_in[0];
  const float* mask  = (const float*)d_in[1];
  const float* Wqkv  = (const float*)d_in[2];
  const float* bqkv  = (const float*)d_in[3];
  const float* Wproj = (const float*)d_in[4];
  const float* bproj = (const float*)d_in[5];
  const float* btab  = (const float*)d_in[6];
  const int*   ridx  = (const int*)d_in[7];
  float* out = (float*)d_out;

  _Float16* wq16 = (_Float16*)d_ws;             // 384*128 halves
  _Float16* wp16 = wq16 + 3 * kC * kC;          // 128*128 halves
  float* biasH   = (float*)(wp16 + kC * kC);    // 4*49*49 floats

  prep_kernel<<<(3 * kC * kC + 255) / 256, 256, 0, stream>>>(
      Wqkv, Wproj, btab, ridx, wq16, wp16, biasH);

  // LDS: x/o 16KB + q/k(=p) 32KB + vT 16KB + s32 64KB = 128 KB -> 2 blocks/WGP
  size_t smem = (size_t)(kNP * kC + 3 * kH * kNP * kD) * 2   // x, q, k, vT (f16)
              + (size_t)(kH * kNP * kNP) * 4;                // s32
  win_attn_kernel<<<kNW, 256, smem, stream>>>(
      x, mask, bqkv, bproj, wq16, wp16, biasH, out);
}
